// A2C2f_60533269070410
// MI455X (gfx1250) — matrix operand order repaired and checked by reading, not compiled, not run on hardware
//
#include <hip/hip_runtime.h>
#include <hip/hip_bf16.h>

// ---------------------------------------------------------------------------
// A2C2f (area-attention C2f) for MI455X / gfx1250, wave32 + WMMA f16.
// Shapes: B=2, C1=C2=512, C_HID=256, H=W=64 (N=4096), HEADS=8, HEAD_DIM=32,
// AREA=4, MLP_H=512, 2 stacks x 2 attention blocks.
// All matmuls use v_wmma_f32_16x16x32_f16 (f16 operands, fp32 accumulate).
// Fast-path epilogues: sigmoid/normalization via v_rcp_f32 instead of the
// IEEE div_scale/div_fmas/div_fixup sequence.
// ---------------------------------------------------------------------------

typedef __attribute__((ext_vector_type(16))) _Float16 v16h;
typedef __attribute__((ext_vector_type(8)))  float    v8f;

#define WMMA_F16(A, B, C) \
  __builtin_amdgcn_wmma_f32_16x16x32_f16(false, (A), false, (B), (short)0, (C), false, false)

__device__ __forceinline__ float fast_rcp(float x) {
  return __builtin_amdgcn_rcpf(x);            // single v_rcp_f32
}
__device__ __forceinline__ float fast_silu(float v) {
  return v * fast_rcp(1.0f + __expf(-v));     // v * sigmoid(v)
}

// ---------------------------------------------------------------------------
// Generic conv1x1 == GEMM: out[b,o,n] = act( sum_c W[o,c]*X[b,c,n] + bias[o] )
// (+ optional residual added after the bias/activation).
// One wave computes a 16(M) x 16(N) tile; K-loop in steps of 32 (1 WMMA each).
// A operand = weight tile (16-bit A layout: lane<16 holds K {0..7,16..23},
// lane>=16 holds K {8..15,24..31} of row M=lane&15).
// B operand = activation tile (lane<16 holds K {0..7,16..23}, lane>=16 K {8..15,24..31} of
// column N=lane&15).
// ---------------------------------------------------------------------------
__global__ __launch_bounds__(128)
void gemm_conv1x1_k(const float* __restrict__ X, const float* __restrict__ W,
                    const float* __restrict__ bias, const float* __restrict__ res,
                    float* __restrict__ out,
                    int K, int act,
                    long in_bstride, long out_bstride, long res_bstride)
{
  const int lane  = threadIdx.x & 31;
  const int wave  = threadIdx.x >> 5;
  const int ntile = blockIdx.x * 4 + wave;   // N=4096 -> 256 tiles
  const int mtile = blockIdx.y;
  const int b     = blockIdx.z;
  const int col   = lane & 15;
  const int hi    = lane >> 4;
  const int n0    = ntile * 16 + col;
  const int m_a   = mtile * 16 + col;        // A-row this lane supplies

  const float* __restrict__ Xb   = X + (long)b * in_bstride;
  const float* __restrict__ wrow = W + (long)m_a * K;

  v8f acc = {};
  for (int kk = 0; kk < K; kk += 32) {
    // A: two contiguous 8-float runs of the weight row (b128-mergeable)
    const int kb = kk + hi * 8;
    v16h a;
#pragma unroll
    for (int i = 0; i < 8; ++i)  a[i]     = (_Float16)wrow[kb + i];
#pragma unroll
    for (int i = 0; i < 8; ++i)  a[8 + i] = (_Float16)wrow[kb + 16 + i];
    // B: activation column n0, K strided by 4096
    const int kb2 = kk + hi * 8;
    v16h bm;
#pragma unroll
    for (int i = 0; i < 16; ++i)
      bm[i] = (_Float16)Xb[(long)(kb2 + (i < 8 ? i : i + 8)) * 4096 + n0];
    acc = WMMA_F16(a, bm, acc);
  }

  float* __restrict__ ob = out + (long)b * out_bstride;
  const float* __restrict__ rb = res ? (res + (long)b * res_bstride) : nullptr;
#pragma unroll
  for (int r = 0; r < 8; ++r) {
    const int m = mtile * 16 + r + hi * 8;   // C/D row for slot r
    float v = acc[r] + bias[m];
    if (act) v = fast_silu(v);
    const long idx = (long)m * 4096 + n0;
    if (rb) v += rb[idx];
    ob[idx] = v;
  }
}

// ---------------------------------------------------------------------------
// Depthwise 5x5 conv with zero padding (positional encoding path).
// ---------------------------------------------------------------------------
__global__ void dwconv5_k(const float* __restrict__ in, const float* __restrict__ w,
                          const float* __restrict__ bias, float* __restrict__ out)
{
  const long i = (long)blockIdx.x * 256 + threadIdx.x;   // 2*256*4096 threads
  if (i >= 2L * 256 * 4096) return;
  const int hw = (int)(i & 4095);
  const int c  = (int)((i >> 12) & 255);
  const int b  = (int)(i >> 20);
  const int h  = hw >> 6, x = hw & 63;
  const float* __restrict__ ib = in + (((long)b * 256 + c) << 12);
  const float* __restrict__ wc = w + c * 25;
  float s = bias[c];
#pragma unroll
  for (int kh = 0; kh < 5; ++kh) {
    const int hh = h + kh - 2;
    if ((unsigned)hh < 64u) {
#pragma unroll
      for (int kw = 0; kw < 5; ++kw) {
        const int ww = x + kw - 2;
        if ((unsigned)ww < 64u) s += ib[hh * 64 + ww] * wc[kh * 5 + kw];
      }
    }
  }
  out[i] = s;
}

// ---------------------------------------------------------------------------
// Flash-style area attention.  The reference reshape (B,C,N)->(B*4,C,N/4)
// splits the CHANNEL axis into 4 area blocks:
//   qk[bn, cr, j] = qk_mem[b, ablk*128 + cr/4, (cr%4)*1024 + j]   (512 ch)
//   v [bn, cr, j] = v_mem [b, ablk*64  + cr/4, (cr%4)*1024 + j]   (256 ch)
// with b = bn>>2, ablk = bn&3, cr = head*32 + d.  q = cr in [0,256),
// k = cr+256.  Per (bn, head, 16-row i-tile) one wave streams j in chunks
// of 32: 2 S-WMMAs (K = HEAD_DIM = 32) -> online softmax -> P via LDS to
// A-layout -> 2 O-WMMAs (K = 32 j's).  O rows are divided by l at the end.
// ---------------------------------------------------------------------------
__global__ __launch_bounds__(128)
void area_attn_k(const float* __restrict__ qk, const float* __restrict__ v,
                 float* __restrict__ out)
{
  __shared__ _Float16 plds[4][16 * 32];
  const int lane  = threadIdx.x & 31;
  const int wave  = threadIdx.x >> 5;
  const int itile = blockIdx.x * 4 + wave;   // 64 i-tiles (n=1024)
  const int head  = blockIdx.y;              // 8 heads
  const int bn    = blockIdx.z;              // 8 = B*AREA
  const int b     = bn >> 2, ablk = bn & 3;
  const int col   = lane & 15;
  const int hi    = lane >> 4;

  // Q A-tile (rows = i, K = d): element e -> d = hi*8 + (e<8 ? e : e+8)
  const int irow = itile * 16 + col;
  v16h aq;
#pragma unroll
  for (int e = 0; e < 16; ++e) {
    const int d  = hi * 8 + (e < 8 ? e : e + 8);
    const int cr = head * 32 + d;                       // q slice: [0,256)
    const long addr = ((long)(b * 512 + ablk * 128 + (cr >> 2)) << 12)
                    + ((cr & 3) << 10) + irow;
    aq[e] = (_Float16)qk[addr];
  }

  v8f O0 = {}, O1 = {};
  float mrow[8], lrow[8];
#pragma unroll
  for (int r = 0; r < 8; ++r) { mrow[r] = -1e30f; lrow[r] = 0.0f; }

  const float scale = 0.17677669529663687f;   // 1/sqrt(32)
  _Float16* __restrict__ pl = plds[wave];

  for (int jj = 0; jj < 1024; jj += 32) {
    // K B-tiles (K = d, cols = j): element e -> d = hi*8 + (e<8 ? e : e+8)
    v16h bk0, bk1;
#pragma unroll
    for (int e = 0; e < 16; ++e) {
      const int d   = hi * 8 + (e < 8 ? e : e + 8);
      const int crk = 256 + head * 32 + d;              // k slice: [256,512)
      const long base = ((long)(b * 512 + ablk * 128 + (crk >> 2)) << 12)
                      + ((crk & 3) << 10) + jj;
      bk0[e] = (_Float16)qk[base + col];
      bk1[e] = (_Float16)qk[base + 16 + col];
    }
    v8f S0 = {}, S1 = {};
    S0 = WMMA_F16(aq, bk0, S0);
    S1 = WMMA_F16(aq, bk1, S1);

    // online softmax: C/D slot r holds row (r + 8*hi), col (lane&15)
    float p0[8], p1[8];
#pragma unroll
    for (int r = 0; r < 8; ++r) { p0[r] = S0[r] * scale; p1[r] = S1[r] * scale; }
#pragma unroll
    for (int r = 0; r < 8; ++r) {
      float t = fmaxf(p0[r], p1[r]);
#pragma unroll
      for (int msk = 1; msk < 16; msk <<= 1) t = fmaxf(t, __shfl_xor(t, msk, 32));
      const float mnew = fmaxf(mrow[r], t);
      const float corr = __expf(mrow[r] - mnew);
      mrow[r] = mnew;
      p0[r] = __expf(p0[r] - mnew);
      p1[r] = __expf(p1[r] - mnew);
      float rs = p0[r] + p1[r];
#pragma unroll
      for (int msk = 1; msk < 16; msk <<= 1) rs += __shfl_xor(rs, msk, 32);
      lrow[r] = lrow[r] * corr + rs;
      O0[r] *= corr;
      O1[r] *= corr;
    }

    // P (16 x 32, f16) to LDS, then reload in A-layout
#pragma unroll
    for (int r = 0; r < 8; ++r) {
      const int row = r + hi * 8;
      pl[row * 32 + col]      = (_Float16)p0[r];
      pl[row * 32 + 16 + col] = (_Float16)p1[r];
    }
    __syncthreads();
    v16h ap;
#pragma unroll
    for (int e = 0; e < 16; ++e) {
      const int k = hi * 16 + e;
      ap[e] = pl[col * 32 + k];
    }
    __syncthreads();

    // V B-tiles (K = j, cols = d): contiguous 16 floats in j per lane
    v16h bv0, bv1;
    {
      const int d0 = col, d1 = 16 + col;
      const long a0 = ((long)(b * 256 + ablk * 64 + ((head * 32 + d0) >> 2)) << 12)
                    + ((d0 & 3) << 10) + jj + hi * 16;
      const long a1 = ((long)(b * 256 + ablk * 64 + ((head * 32 + d1) >> 2)) << 12)
                    + ((d1 & 3) << 10) + jj + hi * 16;
#pragma unroll
      for (int e = 0; e < 16; ++e) {
        bv0[e] = (_Float16)v[a0 + e];
        bv1[e] = (_Float16)v[a1 + e];
      }
    }
    O0 = WMMA_F16(ap, bv0, O0);
    O1 = WMMA_F16(ap, bv1, O1);
  }

  // finalize: divide each row by l (v_rcp_f32), scatter through area mapping
#pragma unroll
  for (int r = 0; r < 8; ++r) {
    const float inv = fast_rcp(lrow[r]);
    const int i = itile * 16 + r + hi * 8;
    const int d0 = col, d1 = 16 + col;
    const long a0 = ((long)(b * 256 + ablk * 64 + ((head * 32 + d0) >> 2)) << 12)
                  + ((d0 & 3) << 10) + i;
    const long a1 = ((long)(b * 256 + ablk * 64 + ((head * 32 + d1) >> 2)) << 12)
                  + ((d1 & 3) << 10) + i;
    out[a0] = O0[r] * inv;
    out[a1] = O1[r] * inv;
  }
}

// ---------------------------------------------------------------------------
// Small elementwise helpers.
// ---------------------------------------------------------------------------
__global__ void add_inplace_k(float* __restrict__ a, const float* __restrict__ b)
{
  const long i = (long)blockIdx.x * 256 + threadIdx.x;
  if (i < 2L * 256 * 4096) a[i] += b[i];
}

__global__ void copy_to_concat_k(const float* __restrict__ src, float* __restrict__ dst)
{
  const long i = (long)blockIdx.x * 256 + threadIdx.x;
  if (i >= 2L * 256 * 4096) return;
  const long n = i & 4095, c = (i >> 12) & 255, b = i >> 20;
  dst[(b * 768 + c) * 4096 + n] = src[i];
}

__global__ void final_residual_k(const float* __restrict__ x, const float* __restrict__ gamma,
                                 const float* __restrict__ t, float* __restrict__ out)
{
  const long i = (long)blockIdx.x * 256 + threadIdx.x;
  if (i >= 2L * 512 * 4096) return;
  const int c = (int)((i >> 12) & 511);
  out[i] = x[i] + gamma[c] * t[i];
}

// ---------------------------------------------------------------------------
// Host orchestration.
// Workspace layout (floats):  concat(2*768*4096) | bufA | bufB | qk | v | pp |
// ao | h(2*512*4096) | t(2*512*4096)  ==> ~117 MB total.
// ---------------------------------------------------------------------------
extern "C" void kernel_launch(void* const* d_in, const int* in_sizes, int n_in,
                              void* d_out, int out_size, void* d_ws, size_t ws_size,
                              hipStream_t stream)
{
  (void)n_in; (void)out_size; (void)ws_size;
  auto F = [&](int i) { return (const float*)d_in[i]; };

  const float *x, *cv1_w, *cv1_b, *cv2_w, *cv2_b, *gamma;
  // canonical per-block order: qk_w,qk_b,v_w,v_b,pe_w,pe_b,proj_w,proj_b,
  //                            mlp1_w,mlp1_b,mlp2_w,mlp2_b
  const float* ab[4][12];

  if (in_sizes[0] == 2 * 512 * 64 * 64) {
    // dict-insertion order: x, cv1_w, cv1_b, cv2_w, cv2_b, gamma, blocks...
    x = F(0); cv1_w = F(1); cv1_b = F(2); cv2_w = F(3); cv2_b = F(4); gamma = F(5);
    for (int blk = 0; blk < 4; ++blk)
      for (int j = 0; j < 12; ++j) ab[blk][j] = F(6 + blk * 12 + j);
  } else {
    // jax pytree (alphabetical) order: blocks first (keys sorted), then
    // cv1_b, cv1_w, cv2_b, cv2_w, gamma, x.
    static const int perm[12] = {9, 8, 11, 10, 5, 4, 7, 6, 1, 0, 3, 2};
    for (int blk = 0; blk < 4; ++blk)
      for (int j = 0; j < 12; ++j) ab[blk][j] = F(blk * 12 + perm[j]);
    cv1_b = F(48); cv1_w = F(49); cv2_b = F(50); cv2_w = F(51); gamma = F(52); x = F(53);
  }

  float* ws     = (float*)d_ws;
  float* concat = ws;                          // 6,291,456
  float* bufA   = concat + 6291456;            // 2,097,152
  float* bufB   = bufA + 2097152;              // 2,097,152
  float* qkb    = bufB + 2097152;              // 4,194,304
  float* vb     = qkb + 4194304;               // 2,097,152
  float* ppb    = vb + 2097152;                // 2,097,152
  float* aob    = ppb + 2097152;               // 2,097,152
  float* hb     = aob + 2097152;               // 4,194,304
  float* tb     = hb + 4194304;                // 4,194,304

  const dim3 blk128(128);
  auto gemm = [&](const float* X, const float* W, const float* B, const float* R,
                  float* O, int K, int Cout, int act, long inb, long outb) {
    dim3 g(64, Cout / 16, 2);
    gemm_conv1x1_k<<<g, blk128, 0, stream>>>(X, W, B, R, O, K, act, inb, outb, outb);
  };

  // y = silu(cv1(x));  ys[0] = y
  gemm(x, cv1_w, cv1_b, nullptr, bufA, 512, 256, 1, 512L * 4096, 256L * 4096);
  copy_to_concat_k<<<8192, 256, 0, stream>>>(bufA, concat);

  for (int s = 0; s < 2; ++s) {
    for (int i = 0; i < 2; ++i) {
      const float* const* p = ab[s * 2 + i];
      // attention branch
      gemm(bufA, p[0], p[1], nullptr, qkb, 256, 512, 0, 256L * 4096, 512L * 4096);
      gemm(bufA, p[2], p[3], nullptr, vb,  256, 256, 0, 256L * 4096, 256L * 4096);
      dwconv5_k<<<8192, 256, 0, stream>>>(vb, p[4], p[5], ppb);
      area_attn_k<<<dim3(16, 8, 8), blk128, 0, stream>>>(qkb, vb, aob);
      add_inplace_k<<<8192, 256, 0, stream>>>(aob, ppb);
      // x1 = x + proj(out + pp)
      gemm(aob, p[6], p[7], bufA, bufB, 256, 256, 0, 256L * 4096, 256L * 4096);
      // x2 = x1 + mlp2(silu(mlp1(x1)))
      gemm(bufB, p[8], p[9],  nullptr, hb,   256, 512, 1, 256L * 4096, 512L * 4096);
      gemm(hb,   p[10], p[11], bufB,   bufA, 512, 256, 0, 512L * 4096, 256L * 4096);
    }
    copy_to_concat_k<<<8192, 256, 0, stream>>>(bufA, concat + (long)(s + 1) * 1048576);
  }

  // out = x + gamma * silu(cv2(concat(ys)))
  gemm(concat, cv2_w, cv2_b, nullptr, tb, 768, 512, 1, 768L * 4096, 512L * 4096);
  final_residual_k<<<16384, 256, 0, stream>>>(x, gamma, tb, (float*)d_out);
}
